// Map_Encoder_12292196401176
// MI455X (gfx1250) — matrix operand order repaired and checked by reading, not compiled, not run on hardware
//
#include <hip/hip_runtime.h>
#include <hip/hip_bf16.h>

typedef __attribute__((ext_vector_type(16))) _Float16 v16h;
typedef __attribute__((ext_vector_type(8)))  _Float16 v8h;
typedef __attribute__((ext_vector_type(8)))  float    v8f;

#define BATCH 32
#define TPTS  1024
#define IMG_H 180
#define IMG_W 330
#define HPAD  217   // H + 37
#define WPAD  367   // W + 37
#define HW2   3690  // 45*82
#define HW2P  3712  // padded to mult of 32
#define FCK   16384

// ---------------------------------------------------------------- utilities
__global__ void k_zero_f32(float* p, int n) {
  int i = blockIdx.x * blockDim.x + threadIdx.x;
  if (i < n) p[i] = 0.0f;
}
__global__ void k_zero_f16(_Float16* p, int n) {
  int i = blockIdx.x * blockDim.x + threadIdx.x;
  if (i < n) p[i] = (_Float16)0.0f;
}
__global__ void k_pack_w(const float* __restrict__ w, _Float16* __restrict__ out,
                         int cout, int kreal, int kpad) {
  int i = blockIdx.x * blockDim.x + threadIdx.x;
  if (i >= cout * kpad) return;
  int co = i / kpad, k = i - co * kpad;
  out[i] = (k < kreal) ? (_Float16)w[co * kreal + k] : (_Float16)0.0f;
}
__global__ void k_pack_f32_f16(const float* __restrict__ in, _Float16* __restrict__ out, int n) {
  int i = blockIdx.x * blockDim.x + threadIdx.x;
  if (i < n) out[i] = (_Float16)in[i];
}
__global__ void k_fold_bn(const float* __restrict__ cb, const float* __restrict__ g,
                          const float* __restrict__ bb, float* __restrict__ sc,
                          float* __restrict__ bi, int c) {
  int i = blockIdx.x * blockDim.x + threadIdx.x;
  if (i >= c) return;
  sc[i] = g[i];
  bi[i] = g[i] * cb[i] + bb[i];
}

// ---------------------------------------------------------------- heatmap
__global__ void k_heat_scatter(const float* __restrict__ xt, float* __restrict__ heat) {
  int idx = blockIdx.x * blockDim.x + threadIdx.x;
  if (idx >= BATCH * TPTS) return;
  int b = idx >> 10;
  float xv = xt[2 * idx], yv = xt[2 * idx + 1];
  if (!(xv == xv) || !(yv == yv)) return;   // NaN check
  int xp = IMG_H - (int)((xv - 30.0f) * 180.0f);
  int yp = (int)((yv - 120.0f) * 165.0f);
  int xs = min(max(xp - 18, 0), IMG_H - 36);
  int xe = min(max(xp + 18, 0), IMG_H);
  int ys = min(max(yp - 18, 0), IMG_W - 36);
  int ye = min(max(yp + 18, 0), IMG_W);
  float k1[37];
  float s = 0.0f;
#pragma unroll
  for (int i = 0; i < 37; i++) {
    float d = (float)(i - 18);
    k1[i] = __expf(-d * d / 18.0f);
    s += k1[i];
  }
  float inv = 1.0f / (s * s);
  int rh = xe - xs, rw = ye - ys;
  for (int r = 0; r < 37; r++) {
    if (r >= rh) break;
    float kr = k1[r] * inv;
    float* row = heat + ((long)b * HPAD + xs + r) * WPAD + ys;
    for (int c = 0; c < 37; c++) {
      if (c >= rw) break;
      atomicAdd(row + c, kr * k1[c]);
    }
  }
}

__global__ void k_heat_fuse(const float* __restrict__ heat, const float* __restrict__ img,
                            _Float16* __restrict__ act0) {
  int b = blockIdx.x, tid = threadIdx.x;
  __shared__ float red[256];
  float mx = -1e30f;
  for (int i = tid; i < IMG_H * IMG_W; i += 256) {
    int y = i / IMG_W, x = i - y * IMG_W;
    mx = fmaxf(mx, heat[((long)b * HPAD + y) * WPAD + x]);
  }
  red[tid] = mx;
  __syncthreads();
  for (int s = 128; s > 0; s >>= 1) {
    if (tid < s) red[tid] = fmaxf(red[tid], red[tid + s]);
    __syncthreads();
  }
  float m = red[0] + 1e-10f;
  for (int i = tid; i < IMG_H * IMG_W; i += 256) {
    int y = i / IMG_W, x = i - y * IMG_W;
    float h = heat[((long)b * HPAD + y) * WPAD + x] / m;
    float f = 0.7f * h + 0.3f * img[(long)b * IMG_H * IMG_W + i];
    act0[(long)b * IMG_H * IMG_W + i] = (_Float16)f;
  }
}

// ------------------------------------------------- implicit-GEMM conv (WMMA)
// One block = COUT/16 waves, all working on the same 16 spatial positions.
// Per 32-wide k-step:
//   - A (weight) panel COUTx32 f16 staged global->LDS via global_load_async_to_lds_b128
//   - B (im2col) tile built cooperatively ONCE into LDS in WMMA fragment layout
//   - each wave: ds_load fragments, v_wmma into its accumulator(s)
// POOL: 4 accumulators for the 2x2 window, max-of-relu epilogue.
template <int CIN, int COUT, int HS, int WS, bool POOL, bool SIGMOID>
__global__ __launch_bounds__(COUT * 2) void k_conv_wmma(
    const _Float16* __restrict__ in, const _Float16* __restrict__ wpack,
    const float* __restrict__ scale, const float* __restrict__ bias,
    _Float16* __restrict__ out, int outStride) {
  constexpr int KREAL = CIN * 9;
  constexpr int KPAD  = ((KREAL + 31) / 32) * 32;
  constexpr int HO    = POOL ? HS / 2 : HS;
  constexpr int WO    = POOL ? WS / 2 : WS;
  constexpr int NPOS  = HO * WO;
  constexpr int NSUB  = POOL ? 4 : 1;
  constexpr int NTHR  = COUT * 2;     // (COUT/16) waves * 32 lanes

  __shared__ alignas(64) _Float16 sA[COUT * 32];   // raw rows: m-th row = 32 halves
  __shared__ alignas(64) _Float16 sB[NSUB * 512];  // fragment layout per sub-tile

  int b  = blockIdx.y;
  int n0 = blockIdx.x * 16;
  int tid  = threadIdx.x;
  int wave = tid >> 5;
  int lane = tid & 31;
  int hi = lane >> 4, lr = lane & 15;
  int m0 = wave * 16;

  int n  = n0 + lr;
  int oy = n / WO, ox = n - oy * WO;

  v8f acc[NSUB];
#pragma unroll
  for (int s = 0; s < NSUB; s++) acc[s] = (v8f){};

  const _Float16* ibase = in + (long)b * CIN * HS * WS;

  for (int k0 = 0; k0 < KPAD; k0 += 32) {
    // ---- stage A panel (COUT x 32 halves) via async copies to LDS ----
    for (int c = tid; c < COUT * 4; c += NTHR) {
      int m = c >> 2, q = c & 3;
      unsigned long long ga =
          (unsigned long long)(size_t)(wpack + (size_t)m * KPAD + k0 + q * 8);
      unsigned lds = (unsigned)(size_t)(sA + m * 32 + q * 8);
      asm volatile("global_load_async_to_lds_b128 %0, %1, off"
                   :: "v"(lds), "v"(ga) : "memory");
    }
    // ---- build im2col B tile(s) once for the whole block ----
    for (int f = tid; f < NSUB * 512; f += NTHR) {
      int sub = f >> 9;
      int r   = f & 511;
      int ln  = r >> 4;          // fragment lane
      int i   = r & 15;          // fragment element
      int khi = ln >> 4, nlr = ln & 15;
      int k  = k0 + khi * 16 + i;
      int nn = n0 + nlr;
      _Float16 v = (_Float16)0.0f;
      if (k < KREAL && nn < NPOS) {
        int yy = nn / WO, xx = nn - yy * WO;
        int py = POOL ? (2 * yy + (sub >> 1)) : yy;
        int px = POOL ? (2 * xx + (sub & 1)) : xx;
        int ci = k / 9;
        int rem = k - ci * 9;
        int dy = rem / 3, dx = rem - dy * 3;
        int iy = py + dy - 1, ix = px + dx - 1;
        if (iy >= 0 && iy < HS && ix >= 0 && ix < WS)
          v = ibase[((long)ci * HS + iy) * WS + ix];
      }
      sB[f] = v;
    }
    asm volatile("s_wait_asynccnt 0" ::: "memory");
    __syncthreads();

    // ---- per-wave fragments + WMMA ----
    v16h af;
    {
      const _Float16* ar = sA + (m0 + lr) * 32 + 16 * hi;
      v8h alo = *(const v8h*)(ar);
      v8h ahi = *(const v8h*)(ar + 8);
#pragma unroll
      for (int i = 0; i < 8; i++) { af[i] = alo[i]; af[i + 8] = ahi[i]; }
    }
#pragma unroll
    for (int sub = 0; sub < NSUB; sub++) {
      v16h bf = *(const v16h*)(sB + sub * 512 + lane * 16);
      acc[sub] = __builtin_amdgcn_wmma_f32_16x16x32_f16(
          false, af, false, bf, (short)0, acc[sub], false, false);
    }
    __syncthreads();
  }

#pragma unroll
  for (int r = 0; r < 8; r++) {
    int m = m0 + r + 8 * hi;
    float sc = scale[m], bi = bias[m];
    float v;
    if (POOL) {
      float a0 = fmaxf(acc[0][r] * sc + bi, 0.0f);
      float a1 = fmaxf(acc[1][r] * sc + bi, 0.0f);
      float a2 = fmaxf(acc[2][r] * sc + bi, 0.0f);
      float a3 = fmaxf(acc[3][r] * sc + bi, 0.0f);
      v = fmaxf(fmaxf(a0, a1), fmaxf(a2, a3));
    } else {
      float t = acc[0][r] * sc + bi;
      v = SIGMOID ? (1.0f / (1.0f + __expf(-t))) : fmaxf(t, 0.0f);
    }
    long obase = ((long)b * COUT + m) * outStride;
    if (n < NPOS) out[obase + n] = (_Float16)v;
    else if (outStride > NPOS && n < outStride) out[obase + n] = (_Float16)0.0f;
  }
}

// ---------------------------------------------------------------- min/max + gate
__global__ void k_minmax(const _Float16* __restrict__ p, float* __restrict__ mm) {
  int b = blockIdx.x, tid = threadIdx.x;
  __shared__ float smn[256], smx[256];
  float mn = 1e30f, mx = -1e30f;
  const _Float16* base = p + (long)b * 128 * HW2;
  for (int i = tid; i < 128 * HW2; i += 256) {
    float v = (float)base[i];
    mn = fminf(mn, v);
    mx = fmaxf(mx, v);
  }
  smn[tid] = mn; smx[tid] = mx;
  __syncthreads();
  for (int s = 128; s > 0; s >>= 1) {
    if (tid < s) {
      smn[tid] = fminf(smn[tid], smn[tid + s]);
      smx[tid] = fmaxf(smx[tid], smx[tid + s]);
    }
    __syncthreads();
  }
  if (tid == 0) { mm[2 * b] = smn[0]; mm[2 * b + 1] = smx[0]; }
}

__global__ void k_pmap(const _Float16* __restrict__ p, const float* __restrict__ mm,
                       _Float16* __restrict__ pmout) {
  int i = blockIdx.x * blockDim.x + threadIdx.x;
  if (i >= BATCH * 128 * HW2) return;
  int b = i / (128 * HW2);
  int r = i - b * 128 * HW2;
  int ch = r / HW2, nn = r - ch * HW2;
  float mn = mm[2 * b], mx = mm[2 * b + 1];
  float v = ((float)p[i] - mn) / fmaxf(mx - mn, 1e-4f);
  v = 1.0f / (1.0f + __expf(-10.0f * (v - 0.5f)));
  pmout[((long)b * 128 + ch) * HW2P + nn] = (_Float16)v;
}

// --------------------------------------------- einsum bphw,bchw->bpc (WMMA)
__global__ __launch_bounds__(32) void k_einsum_wmma(const _Float16* __restrict__ pm,
                                                    const _Float16* __restrict__ fx,
                                                    _Float16* __restrict__ px) {
  int b = blockIdx.y;
  int mt = blockIdx.x >> 3, nt = blockIdx.x & 7;
  int m0 = mt * 16, n0 = nt * 16;
  int lane = threadIdx.x, hi = lane >> 4, lr = lane & 15;
  const _Float16* arow = pm + ((long)b * 128 + m0 + lr) * HW2P + 16 * hi;
  const _Float16* brow = fx + ((long)b * 128 + n0 + lr) * HW2P + 16 * hi;
  v8f c = (v8f){};
  for (int k0 = 0; k0 < HW2P; k0 += 32) {
    v8h alo = *(const v8h*)(arow + k0);
    v8h ahi = *(const v8h*)(arow + k0 + 8);
    v16h af;
#pragma unroll
    for (int i = 0; i < 8; i++) { af[i] = alo[i]; af[i + 8] = ahi[i]; }
    v16h bf = *(const v16h*)(brow + k0);
    c = __builtin_amdgcn_wmma_f32_16x16x32_f16(false, af, false, bf, (short)0, c, false, false);
  }
#pragma unroll
  for (int r = 0; r < 8; r++) {
    int m = m0 + r + 8 * hi;
    px[(long)b * 16384 + m * 128 + n0 + lr] = (_Float16)(c[r] * (1.0f / 3690.0f));
  }
}

// --------------------------------------------------------------- FC (WMMA)
__global__ __launch_bounds__(32) void k_fc_wmma(const _Float16* __restrict__ px,
                                                const _Float16* __restrict__ w,
                                                const float* __restrict__ bias,
                                                float* __restrict__ out) {
  int mt = blockIdx.x >> 4, nt = blockIdx.x & 15;
  int m0 = mt * 16, n0 = nt * 16;
  int lane = threadIdx.x, hi = lane >> 4, lr = lane & 15;
  const _Float16* arow = px + (long)(m0 + lr) * FCK + 16 * hi;
  const _Float16* brow = w + (long)(n0 + lr) * FCK + 16 * hi;
  v8f c = (v8f){};
  for (int k0 = 0; k0 < FCK; k0 += 32) {
    __builtin_prefetch(brow + k0 + 512, 0, 1);
    v8h alo = *(const v8h*)(arow + k0);
    v8h ahi = *(const v8h*)(arow + k0 + 8);
    v16h af;
#pragma unroll
    for (int i = 0; i < 8; i++) { af[i] = alo[i]; af[i + 8] = ahi[i]; }
    v16h bf = *(const v16h*)(brow + k0);
    c = __builtin_amdgcn_wmma_f32_16x16x32_f16(false, af, false, bf, (short)0, c, false, false);
  }
#pragma unroll
  for (int r = 0; r < 8; r++) {
    int m = m0 + r + 8 * hi;
    out[m * 256 + n0 + lr] = c[r] + bias[n0 + lr];
  }
}

// ---------------------------------------------------------------- launcher
extern "C" void kernel_launch(void* const* d_in, const int* in_sizes, int n_in,
                              void* d_out, int out_size, void* d_ws, size_t ws_size,
                              hipStream_t stream) {
  (void)in_sizes; (void)n_in; (void)out_size; (void)ws_size;
  const float* x_t   = (const float*)d_in[0];
  const float* img   = (const float*)d_in[1];
  const float* c1w   = (const float*)d_in[2];
  const float* c1b   = (const float*)d_in[3];
  const float* bn1g  = (const float*)d_in[4];
  const float* bn1b  = (const float*)d_in[5];
  const float* c2w   = (const float*)d_in[6];
  const float* c2b   = (const float*)d_in[7];
  const float* bn2g  = (const float*)d_in[8];
  const float* bn2b  = (const float*)d_in[9];
  const float* a1w   = (const float*)d_in[10];
  const float* a1b   = (const float*)d_in[11];
  const float* abn1g = (const float*)d_in[12];
  const float* abn1b = (const float*)d_in[13];
  const float* a2w   = (const float*)d_in[14];
  const float* a2b   = (const float*)d_in[15];
  const float* abn2g = (const float*)d_in[16];
  const float* abn2b = (const float*)d_in[17];
  const float* p1w   = (const float*)d_in[18];
  const float* p1b   = (const float*)d_in[19];
  const float* pbn1g = (const float*)d_in[20];
  const float* pbn1b = (const float*)d_in[21];
  const float* p2w   = (const float*)d_in[22];
  const float* p2b   = (const float*)d_in[23];
  const float* pbn2g = (const float*)d_in[24];
  const float* pbn2b = (const float*)d_in[25];
  const float* fcw   = (const float*)d_in[26];
  const float* fcb   = (const float*)d_in[27];
  float* out = (float*)d_out;

  char* ws = (char*)d_ws;
  size_t off = 0;
  auto alloc = [&](size_t bytes) -> void* {
    void* p = ws + off;
    off = (off + bytes + 255) & ~(size_t)255;
    return p;
  };

  const int heatN = BATCH * HPAD * WPAD;
  float*    heat  = (float*)alloc((size_t)heatN * 4);
  _Float16* act0  = (_Float16*)alloc((size_t)BATCH * IMG_H * IMG_W * 2);
  _Float16* pool1 = (_Float16*)alloc((size_t)BATCH * 32 * 90 * 165 * 2);
  _Float16* pool2 = (_Float16*)alloc((size_t)BATCH * 64 * HW2 * 2);
  _Float16* fbuf  = (_Float16*)alloc((size_t)BATCH * 128 * HW2 * 2);
  _Float16* fx    = (_Float16*)alloc((size_t)BATCH * 128 * HW2P * 2);
  _Float16* p1buf = (_Float16*)alloc((size_t)BATCH * 128 * HW2 * 2);
  _Float16* p2buf = (_Float16*)alloc((size_t)BATCH * 128 * HW2 * 2);
  _Float16* pmap  = (_Float16*)alloc((size_t)BATCH * 128 * HW2P * 2);
  _Float16* pxbuf = (_Float16*)alloc((size_t)BATCH * 16384 * 2);
  float*    mm    = (float*)alloc(BATCH * 2 * 4);
  _Float16* w1    = (_Float16*)alloc(32 * 32 * 2);
  _Float16* w2    = (_Float16*)alloc(64 * 288 * 2);
  _Float16* wa1   = (_Float16*)alloc(128 * 576 * 2);
  _Float16* wa2   = (_Float16*)alloc(128 * 1152 * 2);
  _Float16* wp1   = (_Float16*)alloc(128 * 576 * 2);
  _Float16* wp2   = (_Float16*)alloc(128 * 1152 * 2);
  _Float16* wfc   = (_Float16*)alloc((size_t)256 * FCK * 2);
  float* sc1 = (float*)alloc(32 * 4);  float* bi1 = (float*)alloc(32 * 4);
  float* sc2 = (float*)alloc(64 * 4);  float* bi2 = (float*)alloc(64 * 4);
  float* sa1 = (float*)alloc(128 * 4); float* ba1 = (float*)alloc(128 * 4);
  float* sa2 = (float*)alloc(128 * 4); float* ba2 = (float*)alloc(128 * 4);
  float* sp1 = (float*)alloc(128 * 4); float* bp1 = (float*)alloc(128 * 4);
  float* sp2 = (float*)alloc(128 * 4); float* bp2 = (float*)alloc(128 * 4);

  auto cdiv = [](int a, int b) { return (a + b - 1) / b; };

  // zero scatter target + padded buffers
  k_zero_f32<<<cdiv(heatN, 256), 256, 0, stream>>>(heat, heatN);
  const int padN = BATCH * 128 * HW2P;
  k_zero_f16<<<cdiv(padN, 256), 256, 0, stream>>>(fx, padN);
  k_zero_f16<<<cdiv(padN, 256), 256, 0, stream>>>(pmap, padN);

  // weight packing + BN folding
  k_pack_w<<<cdiv(32 * 32, 256), 256, 0, stream>>>(c1w, w1, 32, 9, 32);
  k_pack_w<<<cdiv(64 * 288, 256), 256, 0, stream>>>(c2w, w2, 64, 288, 288);
  k_pack_w<<<cdiv(128 * 576, 256), 256, 0, stream>>>(a1w, wa1, 128, 576, 576);
  k_pack_w<<<cdiv(128 * 1152, 256), 256, 0, stream>>>(a2w, wa2, 128, 1152, 1152);
  k_pack_w<<<cdiv(128 * 576, 256), 256, 0, stream>>>(p1w, wp1, 128, 576, 576);
  k_pack_w<<<cdiv(128 * 1152, 256), 256, 0, stream>>>(p2w, wp2, 128, 1152, 1152);
  k_pack_f32_f16<<<cdiv(256 * FCK, 256), 256, 0, stream>>>(fcw, wfc, 256 * FCK);
  k_fold_bn<<<1, 128, 0, stream>>>(c1b, bn1g, bn1b, sc1, bi1, 32);
  k_fold_bn<<<1, 128, 0, stream>>>(c2b, bn2g, bn2b, sc2, bi2, 64);
  k_fold_bn<<<1, 128, 0, stream>>>(a1b, abn1g, abn1b, sa1, ba1, 128);
  k_fold_bn<<<1, 128, 0, stream>>>(a2b, abn2g, abn2b, sa2, ba2, 128);
  k_fold_bn<<<1, 128, 0, stream>>>(p1b, pbn1g, pbn1b, sp1, bp1, 128);
  k_fold_bn<<<1, 128, 0, stream>>>(p2b, pbn2g, pbn2b, sp2, bp2, 128);

  // heatmap + fuse
  k_heat_scatter<<<cdiv(BATCH * TPTS, 256), 256, 0, stream>>>(x_t, heat);
  k_heat_fuse<<<BATCH, 256, 0, stream>>>(heat, img, act0);

  // conv stack (implicit-GEMM WMMA, LDS-shared B, async-staged A)
  {
    const int NT1 = cdiv(90 * 165, 16);
    k_conv_wmma<1, 32, 180, 330, true, false>
        <<<dim3(NT1, BATCH), 64, 0, stream>>>(act0, w1, sc1, bi1, pool1, 90 * 165);
    const int NT = cdiv(HW2, 16);
    k_conv_wmma<32, 64, 90, 165, true, false>
        <<<dim3(NT, BATCH), 128, 0, stream>>>(pool1, w2, sc2, bi2, pool2, HW2);
    k_conv_wmma<64, 128, 45, 82, false, false>
        <<<dim3(NT, BATCH), 256, 0, stream>>>(pool2, wa1, sa1, ba1, fbuf, HW2);
    k_conv_wmma<128, 128, 45, 82, false, true>
        <<<dim3(NT, BATCH), 256, 0, stream>>>(fbuf, wa2, sa2, ba2, fx, HW2P);
    k_conv_wmma<64, 128, 45, 82, false, false>
        <<<dim3(NT, BATCH), 256, 0, stream>>>(pool2, wp1, sp1, bp1, p1buf, HW2);
    k_conv_wmma<128, 128, 45, 82, false, false>
        <<<dim3(NT, BATCH), 256, 0, stream>>>(p1buf, wp2, sp2, bp2, p2buf, HW2);
  }

  // attention-map normalize + gate
  k_minmax<<<BATCH, 256, 0, stream>>>(p2buf, mm);
  k_pmap<<<cdiv(BATCH * 128 * HW2, 256), 256, 0, stream>>>(p2buf, mm, pmap);

  // batched 128x128 einsum GEMM + final FC GEMM
  k_einsum_wmma<<<dim3(64, BATCH), 32, 0, stream>>>(pmap, fx, pxbuf);
  k_fc_wmma<<<32, 32, 0, stream>>>(pxbuf, wfc, fcb, out);
}
